// Model_62345745269552
// MI455X (gfx1250) — matrix operand order repaired and checked by reading, not compiled, not run on hardware
//
#include <hip/hip_runtime.h>

// ---------------- problem constants ----------------
#define NX      1024
#define NHEAD   16
#define HD      64          // head dim
#define BATCH   4
#define SEQ     2048
#define MROWS   (BATCH*SEQ)             // 8192
#define A_ELEMS ((size_t)MROWS*NX)      // 8388608  (output 'a')
#define KV_ELEMS ((size_t)BATCH*NHEAD*SEQ*HD) // 8388608 per tensor
#define WA_ELEMS ((size_t)NX*3*NX)      // 3145728
#define WP_ELEMS ((size_t)NX*NX)        // 1048576

typedef __bf16 bf16_t;
typedef __attribute__((ext_vector_type(16))) __bf16 v16bf;
typedef __attribute__((ext_vector_type(8)))  __bf16 v8bf;
typedef __attribute__((ext_vector_type(4)))  __bf16 v4bf;
typedef __attribute__((ext_vector_type(8)))  float  v8f;
typedef __attribute__((ext_vector_type(4)))  float  v4f;
typedef __attribute__((ext_vector_type(4)))  unsigned int u32x4;
typedef __attribute__((ext_vector_type(4)))  int i32x4;
typedef __attribute__((ext_vector_type(8)))  int i32x8;

#define DEV __device__ __forceinline__

#if defined(__has_builtin)
#if __has_builtin(__builtin_amdgcn_tensor_load_to_lds) && __has_builtin(__builtin_amdgcn_s_wait_tensorcnt)
#define HAVE_TDM 1
#endif
#endif

#ifdef HAVE_TDM
#if __clang_major__ >= 23
#define TDM_CALL(g0, g1) __builtin_amdgcn_tensor_load_to_lds((g0), (g1), \
    (i32x4){0,0,0,0}, (i32x4){0,0,0,0}, (i32x8){0,0,0,0,0,0,0,0}, 0)
#else
#define TDM_CALL(g0, g1) __builtin_amdgcn_tensor_load_to_lds((g0), (g1), \
    (i32x4){0,0,0,0}, (i32x4){0,0,0,0}, 0)
#endif

// 2D tile load: bf16 elements, row-major tensor, LDS padding via pad_enable.
// pad_int_code: row bytes = 8 << code ; pad_amt_code: pad = (code+1) DWORDs.
DEV void tdm_issue_2d(unsigned lds_off, const void* gaddr,
                      unsigned tile_x, unsigned tile_rows,
                      unsigned tensor_row_elems, unsigned tensor_rows,
                      unsigned long long row_stride_elems,
                      unsigned pad_int_code, unsigned pad_amt_code)
{
    unsigned long long ga = (unsigned long long)gaddr;
    u32x4 g0;
    g0[0] = 1u;                                           // count=1, user mode
    g0[1] = lds_off;                                      // lds_addr
    g0[2] = (unsigned)ga;                                 // global_addr[31:0]
    g0[3] = (unsigned)((ga >> 32) & 0x01ffffffu) | (2u << 30); // addr[56:32], type=2
    i32x8 g1;
    g1[0] = (int)((1u << 16)                              // data_size = 2 bytes
                | (1u << 20)                              // pad_enable
                | (pad_int_code << 22) | (pad_amt_code << 25));
    g1[1] = (int)((tensor_row_elems & 0xffffu) << 16);            // tensor_dim0[15:0]
    g1[2] = (int)(((tensor_row_elems >> 16) & 0xffffu)
                | ((tensor_rows & 0xffffu) << 16));               // dim0 hi | dim1 lo
    g1[3] = (int)(((tensor_rows >> 16) & 0xffffu)
                | ((tile_x & 0xffffu) << 16));                    // dim1 hi | tile_dim0
    g1[4] = (int)(tile_rows & 0xffffu);                           // tile_dim1, tile_dim2=0
    g1[5] = (int)(unsigned)(row_stride_elems & 0xffffffffull);    // dim0_stride lo
    g1[6] = (int)(unsigned)((row_stride_elems >> 32) & 0xffffull);// dim0_stride hi
    g1[7] = 0;
    TDM_CALL(g0, g1);
}

DEV unsigned lds_off32(const void* p) { return (unsigned)(unsigned long long)(uintptr_t)p; }
#endif // HAVE_TDM

DEV v16bf cat8(v8bf lo, v8bf hi) {
    return __builtin_shufflevector(lo, hi, 0,1,2,3,4,5,6,7,8,9,10,11,12,13,14,15);
}

DEV v8f wmma_bf16(v16bf a, v16bf b, v8f c) {
    return __builtin_amdgcn_wmma_f32_16x16x32_bf16(false, a, false, b, (short)0, c, false, false);
}

// A-frag (16x32, row per lane): lanes 0-15 hold K 0..7 & 16..23, lanes 16-31 hold 8..15 & 24..31
DEV v16bf load_a_frag(const bf16_t* base, int row, int ld, int kb, int lane) {
    int off = row * ld + kb + ((lane >> 4) << 3);
    v8bf lo = *(const v8bf*)(base + off);
    v8bf hi = *(const v8bf*)(base + off + 16);
    return cat8(lo, hi);
}

// B-frag (32x16): lane = N column (lane & 15); lanes 0-15 hold K 0..7 & 16..23, lanes 16-31 hold 8..15 & 24..31
DEV v16bf load_b_frag(const bf16_t* base, int kb, int nb, int ld, int lane) {
    int off = (kb + ((lane >> 4) << 3)) * ld + nb + (lane & 15);
    v16bf r;
    #pragma unroll
    for (int s = 0; s < 16; ++s) r[s] = base[off + (s < 8 ? s : s + 8) * ld];
    return r;
}

// ============================================================
// Kernel 0: f32 -> bf16 conversion (vectorized, one shot)
// ============================================================
__global__ __launch_bounds__(256)
void cvt_f32_bf16(const float* __restrict__ s, bf16_t* __restrict__ d, int n4)
{
    const int i = blockIdx.x * blockDim.x + threadIdx.x;
    if (i < n4) {
        v4f f = *(const v4f*)(s + (size_t)i * 4);
        v4bf o;
        #pragma unroll
        for (int e = 0; e < 4; ++e) o[e] = (bf16_t)f[e];
        *(v4bf*)(d + (size_t)i * 4) = o;
    }
}

// ============================================================
// Kernel 1: qkv = x @ W_attn + b_attn   (8192x1024 * 1024x3072), bf16 inputs.
// TDM double-buffered LDS pipeline; writes q/k/v bf16 ws and f32 present.
// ============================================================
#define LDA_P 40     // 32 + 8 pad (TDM pad: 64B rows + 16B)
#define LDB_P 136    // 128 + 8 pad (TDM pad: 256B rows + 16B)
#define NIT   (NX / 32)

__global__ __launch_bounds__(256)
void qkv_gemm(const bf16_t* __restrict__ X, const bf16_t* __restrict__ W,
              const float* __restrict__ bias,
              bf16_t* __restrict__ qws, bf16_t* __restrict__ kws, bf16_t* __restrict__ vws,
              float* __restrict__ out)
{
    __shared__ __attribute__((aligned(16))) bf16_t As[2][128 * LDA_P];
    __shared__ __attribute__((aligned(16))) bf16_t Bs[2][32 * LDB_P];

    const int tid   = threadIdx.x;
    const int lane  = tid & 31;
    const int wave  = tid >> 5;
    const int waveM = wave & 3;
    const int waveN = wave >> 2;
    const int blockM = blockIdx.x * 128;
    const int blockN = blockIdx.y * 128;

    v8f acc[8];
    #pragma unroll
    for (int i = 0; i < 8; ++i) acc[i] = (v8f){0,0,0,0,0,0,0,0};

#ifdef HAVE_TDM
    const bf16_t* Abase = X + (size_t)blockM * NX;
    const bf16_t* Bbase = W + blockN;
    const unsigned aoff0 = lds_off32(&As[0][0]), aoff1 = lds_off32(&As[1][0]);
    const unsigned boff0 = lds_off32(&Bs[0][0]), boff1 = lds_off32(&Bs[1][0]);

    if (wave == 0) {
        tdm_issue_2d(aoff0, Abase, 32, 128, NX, MROWS, NX, 3, 3);
        tdm_issue_2d(boff0, Bbase, 128, 32, 3 * NX, NX, 3 * NX, 5, 3);
    }
    for (int i = 0; i < NIT; ++i) {
        const int cur = i & 1;
        if (wave == 0) __builtin_amdgcn_s_wait_tensorcnt(0);
        __syncthreads();
        if (wave == 0 && i + 1 < NIT) {
            tdm_issue_2d(cur ? aoff0 : aoff1, Abase + (i + 1) * 32,
                         32, 128, NX, MROWS, NX, 3, 3);
            tdm_issue_2d(cur ? boff0 : boff1, Bbase + (size_t)(i + 1) * 32 * (3 * NX),
                         128, 32, 3 * NX, NX, 3 * NX, 5, 3);
        }
        const bf16_t* Ab = As[cur];
        const bf16_t* Bb = Bs[cur];
        v16bf aF[2], bF[4];
        #pragma unroll
        for (int mi = 0; mi < 2; ++mi)
            aF[mi] = load_a_frag(Ab, waveM * 32 + mi * 16 + (lane & 15), LDA_P, 0, lane);
        #pragma unroll
        for (int ni = 0; ni < 4; ++ni)
            bF[ni] = load_b_frag(Bb, 0, waveN * 64 + ni * 16, LDB_P, lane);
        #pragma unroll
        for (int mi = 0; mi < 2; ++mi)
            #pragma unroll
            for (int ni = 0; ni < 4; ++ni)
                acc[mi * 4 + ni] = wmma_bf16(aF[mi], bF[ni], acc[mi * 4 + ni]);
    }
#else
    for (int k0 = 0; k0 < NX; k0 += 32) {
        __syncthreads();
        {   // stage A: 128x32 bf16
            const int col8 = tid & 3, row0 = tid >> 2;
            #pragma unroll
            for (int rr = 0; rr < 2; ++rr) {
                const int row = row0 + rr * 64;
                v8bf v = *(const v8bf*)(X + (size_t)(blockM + row) * NX + k0 + col8 * 8);
                *(v8bf*)(&As[0][0] + row * LDA_P + col8 * 8) = v;
            }
        }
        {   // stage B: 32x128 bf16
            const int col8 = tid & 15, row0 = tid >> 4;
            #pragma unroll
            for (int rr = 0; rr < 2; ++rr) {
                const int row = row0 + rr * 16;
                v8bf v = *(const v8bf*)(W + (size_t)(k0 + row) * (3 * NX) + blockN + col8 * 8);
                *(v8bf*)(&Bs[0][0] + row * LDB_P + col8 * 8) = v;
            }
        }
        __syncthreads();
        v16bf aF[2], bF[4];
        #pragma unroll
        for (int mi = 0; mi < 2; ++mi)
            aF[mi] = load_a_frag(&As[0][0], waveM * 32 + mi * 16 + (lane & 15), LDA_P, 0, lane);
        #pragma unroll
        for (int ni = 0; ni < 4; ++ni)
            bF[ni] = load_b_frag(&Bs[0][0], 0, waveN * 64 + ni * 16, LDB_P, lane);
        #pragma unroll
        for (int mi = 0; mi < 2; ++mi)
            #pragma unroll
            for (int ni = 0; ni < 4; ++ni)
                acc[mi * 4 + ni] = wmma_bf16(aF[mi], bF[ni], acc[mi * 4 + ni]);
    }
#endif

    // epilogue: route to q/k/v [B,H,S,D]; whole block lies in one segment
    const int seg = blockN >> 10;  // 0=q 1=k 2=v
    #pragma unroll
    for (int mi = 0; mi < 2; ++mi) {
        const int rowBase = blockM + waveM * 32 + mi * 16 + ((lane >> 4) << 3);
        #pragma unroll
        for (int ni = 0; ni < 4; ++ni) {
            const int col = blockN + waveN * 64 + ni * 16 + (lane & 15);
            const float bv = bias[col];
            const int c1 = col - (seg << 10);
            const int h = c1 >> 6, d = c1 & 63;
            #pragma unroll
            for (int r = 0; r < 8; ++r) {
                const int row = rowBase + r;                 // b*S + s
                const int b = row >> 11, s = row & 2047;
                const float val = acc[mi * 4 + ni][r] + bv;
                const size_t idx = (((size_t)(b * NHEAD + h)) * SEQ + s) * HD + d;
                if (seg == 0)      { qws[idx] = (bf16_t)val; }
                else if (seg == 1) { kws[idx] = (bf16_t)val; out[A_ELEMS + idx] = val; }
                else               { vws[idx] = (bf16_t)val; out[A_ELEMS + KV_ELEMS + idx] = val; }
            }
        }
    }
}

// ============================================================
// Kernel 2: causal flash attention, no 1/sqrt(d) scale.
// grid (S/64, B*H), 128 threads (4 waves x 16 query rows)
// ============================================================
#define LDT 72   // 64 + 8 pad

__global__ __launch_bounds__(128)
void attn_kernel(const bf16_t* __restrict__ qg, const bf16_t* __restrict__ kg,
                 const bf16_t* __restrict__ vg, bf16_t* __restrict__ attn)
{
    __shared__ __attribute__((aligned(16))) bf16_t KT[64 * LDT];        // [d][key]
    __shared__ __attribute__((aligned(16))) bf16_t Vs[64 * LDT];        // [key][d]
    __shared__ __attribute__((aligned(16))) bf16_t Ps[4][16 * LDT];     // per-wave P

    const int tid  = threadIdx.x;
    const int lane = tid & 31;
    const int wave = tid >> 5;
    const int q0   = blockIdx.x * 64;
    const int bh   = blockIdx.y;
    const int b    = bh >> 4, h = bh & 15;

    const bf16_t* qp = qg + (size_t)bh * SEQ * HD;
    const bf16_t* kp = kg + (size_t)bh * SEQ * HD;
    const bf16_t* vp = vg + (size_t)bh * SEQ * HD;

    const int qrow = q0 + wave * 16 + (lane & 15);
    v16bf aQ[2];
    #pragma unroll
    for (int t = 0; t < 2; ++t) {
        const int off = qrow * HD + t * 32 + ((lane >> 4) << 3);
        aQ[t] = cat8(*(const v8bf*)(qp + off), *(const v8bf*)(qp + off + 16));
    }

    v8f o[4];
    #pragma unroll
    for (int i = 0; i < 4; ++i) o[i] = (v8f){0,0,0,0,0,0,0,0};
    float m_i[8], l_i[8];
    #pragma unroll
    for (int r = 0; r < 8; ++r) { m_i[r] = -1e30f; l_i[r] = 0.0f; }

    const int nkt = blockIdx.x + 1;
    for (int kt = 0; kt < nkt; ++kt) {
        const int kbase = kt * 64;
        __syncthreads();
        {   // stage K^T (transposed) and V
            const int key0 = tid & 63, dh = tid >> 6;
            #pragma unroll
            for (int j = 0; j < 4; ++j) {
                v8bf kv = *(const v8bf*)(kp + (size_t)(kbase + key0) * HD + dh * 32 + j * 8);
                #pragma unroll
                for (int e = 0; e < 8; ++e)
                    KT[(dh * 32 + j * 8 + e) * LDT + key0] = kv[e];
                v8bf vv = *(const v8bf*)(vp + (size_t)(kbase + key0) * HD + dh * 32 + j * 8);
                *(v8bf*)(Vs + key0 * LDT + dh * 32 + j * 8) = vv;
            }
        }
        __syncthreads();

        v8f s4[4];
        #pragma unroll
        for (int i = 0; i < 4; ++i) s4[i] = (v8f){0,0,0,0,0,0,0,0};
        #pragma unroll
        for (int t = 0; t < 2; ++t) {
            #pragma unroll
            for (int ni = 0; ni < 4; ++ni) {
                v16bf bK = load_b_frag(KT, t * 32, ni * 16, LDT, lane);
                s4[ni] = wmma_bf16(aQ[t], bK, s4[ni]);
            }
        }

        if (kt == blockIdx.x) {   // diagonal tile: causal mask
            #pragma unroll
            for (int ni = 0; ni < 4; ++ni) {
                const int kj = kbase + ni * 16 + (lane & 15);
                #pragma unroll
                for (int r = 0; r < 8; ++r) {
                    const int qi = q0 + wave * 16 + r + ((lane >> 4) << 3);
                    if (kj > qi) s4[ni][r] = -1e30f;
                }
            }
        }

        #pragma unroll
        for (int r = 0; r < 8; ++r) {
            float t = fmaxf(fmaxf(s4[0][r], s4[1][r]), fmaxf(s4[2][r], s4[3][r]));
            #pragma unroll
            for (int mk = 1; mk < 16; mk <<= 1) t = fmaxf(t, __shfl_xor(t, mk, 16));
            const float mnew  = fmaxf(m_i[r], t);
            const float alpha = __expf(m_i[r] - mnew);
            m_i[r] = mnew;
            float rs = 0.0f;
            #pragma unroll
            for (int ni = 0; ni < 4; ++ni) {
                const float p = __expf(s4[ni][r] - mnew);
                s4[ni][r] = p;
                rs += p;
                o[ni][r] *= alpha;
            }
            #pragma unroll
            for (int mk = 1; mk < 16; mk <<= 1) rs += __shfl_xor(rs, mk, 16);
            l_i[r] = l_i[r] * alpha + rs;
        }

        bf16_t* pw = &Ps[wave][0];
        #pragma unroll
        for (int ni = 0; ni < 4; ++ni)
            #pragma unroll
            for (int r = 0; r < 8; ++r)
                pw[(r + ((lane >> 4) << 3)) * LDT + ni * 16 + (lane & 15)] = (bf16_t)s4[ni][r];

        #pragma unroll
        for (int t = 0; t < 2; ++t) {
            v16bf aP = load_a_frag(pw, lane & 15, LDT, t * 32, lane);
            #pragma unroll
            for (int ni = 0; ni < 4; ++ni) {
                v16bf bV = load_b_frag(Vs, t * 32, ni * 16, LDT, lane);
                o[ni] = wmma_bf16(aP, bV, o[ni]);
            }
        }
    }

    #pragma unroll
    for (int r = 0; r < 8; ++r) {
        const float inv = 1.0f / l_i[r];
        const int qi = q0 + wave * 16 + r + ((lane >> 4) << 3);
        #pragma unroll
        for (int ni = 0; ni < 4; ++ni) {
            const int d = ni * 16 + (lane & 15);
            attn[((size_t)(b * SEQ + qi)) * NX + h * HD + d] = (bf16_t)(o[ni][r] * inv);
        }
    }
}

// ============================================================
// Kernel 3: a = attn @ W_proj + b_proj  (8192x1024 * 1024x1024) -> f32 d_out
// ============================================================
__global__ __launch_bounds__(256)
void proj_gemm(const bf16_t* __restrict__ Aattn, const bf16_t* __restrict__ W,
               const float* __restrict__ bias, float* __restrict__ out)
{
    __shared__ __attribute__((aligned(16))) bf16_t As[2][128 * LDA_P];
    __shared__ __attribute__((aligned(16))) bf16_t Bs[2][32 * LDB_P];

    const int tid   = threadIdx.x;
    const int lane  = tid & 31;
    const int wave  = tid >> 5;
    const int waveM = wave & 3;
    const int waveN = wave >> 2;
    const int blockM = blockIdx.x * 128;
    const int blockN = blockIdx.y * 128;

    v8f acc[8];
    #pragma unroll
    for (int i = 0; i < 8; ++i) acc[i] = (v8f){0,0,0,0,0,0,0,0};

#ifdef HAVE_TDM
    const bf16_t* Abase = Aattn + (size_t)blockM * NX;
    const bf16_t* Bbase = W + blockN;
    const unsigned aoff0 = lds_off32(&As[0][0]), aoff1 = lds_off32(&As[1][0]);
    const unsigned boff0 = lds_off32(&Bs[0][0]), boff1 = lds_off32(&Bs[1][0]);

    if (wave == 0) {
        tdm_issue_2d(aoff0, Abase, 32, 128, NX, MROWS, NX, 3, 3);
        tdm_issue_2d(boff0, Bbase, 128, 32, NX, NX, NX, 5, 3);
    }
    for (int i = 0; i < NIT; ++i) {
        const int cur = i & 1;
        if (wave == 0) __builtin_amdgcn_s_wait_tensorcnt(0);
        __syncthreads();
        if (wave == 0 && i + 1 < NIT) {
            tdm_issue_2d(cur ? aoff0 : aoff1, Abase + (i + 1) * 32,
                         32, 128, NX, MROWS, NX, 3, 3);
            tdm_issue_2d(cur ? boff0 : boff1, Bbase + (size_t)(i + 1) * 32 * NX,
                         128, 32, NX, NX, NX, 5, 3);
        }
        const bf16_t* Ab = As[cur];
        const bf16_t* Bb = Bs[cur];
        v16bf aF[2], bF[4];
        #pragma unroll
        for (int mi = 0; mi < 2; ++mi)
            aF[mi] = load_a_frag(Ab, waveM * 32 + mi * 16 + (lane & 15), LDA_P, 0, lane);
        #pragma unroll
        for (int ni = 0; ni < 4; ++ni)
            bF[ni] = load_b_frag(Bb, 0, waveN * 64 + ni * 16, LDB_P, lane);
        #pragma unroll
        for (int mi = 0; mi < 2; ++mi)
            #pragma unroll
            for (int ni = 0; ni < 4; ++ni)
                acc[mi * 4 + ni] = wmma_bf16(aF[mi], bF[ni], acc[mi * 4 + ni]);
    }
#else
    for (int k0 = 0; k0 < NX; k0 += 32) {
        __syncthreads();
        {
            const int col8 = tid & 3, row0 = tid >> 2;
            #pragma unroll
            for (int rr = 0; rr < 2; ++rr) {
                const int row = row0 + rr * 64;
                v8bf v = *(const v8bf*)(Aattn + (size_t)(blockM + row) * NX + k0 + col8 * 8);
                *(v8bf*)(&As[0][0] + row * LDA_P + col8 * 8) = v;
            }
        }
        {
            const int col8 = tid & 15, row0 = tid >> 4;
            #pragma unroll
            for (int rr = 0; rr < 2; ++rr) {
                const int row = row0 + rr * 16;
                v8bf v = *(const v8bf*)(W + (size_t)(k0 + row) * NX + blockN + col8 * 8);
                *(v8bf*)(&Bs[0][0] + row * LDB_P + col8 * 8) = v;
            }
        }
        __syncthreads();
        v16bf aF[2], bF[4];
        #pragma unroll
        for (int mi = 0; mi < 2; ++mi)
            aF[mi] = load_a_frag(&As[0][0], waveM * 32 + mi * 16 + (lane & 15), LDA_P, 0, lane);
        #pragma unroll
        for (int ni = 0; ni < 4; ++ni)
            bF[ni] = load_b_frag(&Bs[0][0], 0, waveN * 64 + ni * 16, LDB_P, lane);
        #pragma unroll
        for (int mi = 0; mi < 2; ++mi)
            #pragma unroll
            for (int ni = 0; ni < 4; ++ni)
                acc[mi * 4 + ni] = wmma_bf16(aF[mi], bF[ni], acc[mi * 4 + ni]);
    }
#endif

    #pragma unroll
    for (int mi = 0; mi < 2; ++mi) {
        const int rowBase = blockM + waveM * 32 + mi * 16 + ((lane >> 4) << 3);
        #pragma unroll
        for (int ni = 0; ni < 4; ++ni) {
            const int col = blockN + waveN * 64 + ni * 16 + (lane & 15);
            const float bv = bias[col];
            #pragma unroll
            for (int r = 0; r < 8; ++r)
                out[(size_t)(rowBase + r) * NX + col] = acc[mi * 4 + ni][r] + bv;
        }
    }
}

// ============================================================
extern "C" void kernel_launch(void* const* d_in, const int* in_sizes, int n_in,
                              void* d_out, int out_size, void* d_ws, size_t ws_size,
                              hipStream_t stream)
{
    const float* x      = (const float*)d_in[0];
    const float* w_attn = (const float*)d_in[1];
    const float* b_attn = (const float*)d_in[2];
    const float* w_proj = (const float*)d_in[3];
    const float* b_proj = (const float*)d_in[4];
    float* out = (float*)d_out;

    bf16_t* qws = (bf16_t*)d_ws;
    bf16_t* kws = qws + KV_ELEMS;
    bf16_t* vws = kws + KV_ELEMS;
    bf16_t* aws = vws + KV_ELEMS;
    bf16_t* xb  = aws + A_ELEMS;
    bf16_t* wab = xb  + A_ELEMS;
    bf16_t* wpb = wab + WA_ELEMS;   // total ws use ~92 MB

    // one-shot f32 -> bf16 conversions
    cvt_f32_bf16<<<(int)(A_ELEMS / 4 / 256), 256, 0, stream>>>(x, xb, (int)(A_ELEMS / 4));
    cvt_f32_bf16<<<(int)(WA_ELEMS / 4 / 256), 256, 0, stream>>>(w_attn, wab, (int)(WA_ELEMS / 4));
    cvt_f32_bf16<<<(int)(WP_ELEMS / 4 / 256), 256, 0, stream>>>(w_proj, wpb, (int)(WP_ELEMS / 4));

    qkv_gemm<<<dim3(MROWS / 128, (3 * NX) / 128), 256, 0, stream>>>(
        xb, wab, b_attn, qws, kws, vws, out);
    attn_kernel<<<dim3(SEQ / 64, BATCH * NHEAD), 128, 0, stream>>>(
        qws, kws, vws, aws);
    proj_gemm<<<dim3(MROWS / 128, NX / 128), 256, 0, stream>>>(
        aws, wpb, b_proj, out);
}
